// _FCtL_7430293422244
// MI455X (gfx1250) — compile-verified, matrix operand order checked by reading, not run on hardware
//
#include <hip/hip_runtime.h>
#include <hip/hip_bf16.h>

typedef __bf16 bf16_t;
typedef __bf16 bf16x16 __attribute__((ext_vector_type(16)));
typedef float  f32x8   __attribute__((ext_vector_type(8)));
typedef int    i32x4   __attribute__((ext_vector_type(4)));

static __device__ __forceinline__ f32x8 zero8() {
    f32x8 v = {0.f,0.f,0.f,0.f,0.f,0.f,0.f,0.f};
    return v;
}

// Assemble a 16-element bf16 A/B fragment from two 16-byte chunks.
// Layout (per ISA 16-bit A/B 16x32): elems 0..7 = "lo" K-chunk (VGPR0..3),
// elems 8..15 = "hi" K-chunk (VGPR4..7).
static __device__ __forceinline__ bf16x16 load_frag(const bf16_t* lo, const bf16_t* hi) {
    union { bf16x16 v; i32x4 q[2]; } f;
    f.q[0] = *(const i32x4*)lo;
    f.q[1] = *(const i32x4*)hi;
    return f.v;
}

// ---------------------------------------------------------------------------
// f32 -> bf16 conversion
// ---------------------------------------------------------------------------
__global__ void cvt_kernel(const float* __restrict__ src, bf16_t* __restrict__ dst, int n) {
    int i = blockIdx.x * blockDim.x + threadIdx.x;
    if (i < n) dst[i] = (bf16_t)src[i];
}

// ---------------------------------------------------------------------------
// Generic bf16 WMMA GEMM:  Out[b][m][n] = sum_k W[m][k] * X[b][k][n] (+ bias[m])
// 128x128 workgroup tile, 8 waves arranged 4(m) x 2(n); each wave owns 32m x 64n.
// Optional outputs: f32 natural, bf16 natural, bf16 transposed [b][n][m].
// ---------------------------------------------------------------------------
__global__ __launch_bounds__(256) void wmma_gemm(
    const bf16_t* __restrict__ Wm, const bf16_t* __restrict__ X,
    const float* __restrict__ bias,
    float* __restrict__ outF, bf16_t* __restrict__ outB, bf16_t* __restrict__ outT,
    int M, int Ntot, int K,
    int xRows, int xRowOff, int oRows, int oRowOff)
{
    __shared__ bf16_t sW[128][40];   // [m][k], padded stride 40 (80B, 16B-aligned rows)
    __shared__ bf16_t sX[128][40];   // [n][k] (transposed on fill)

    const int t    = threadIdx.x;
    const int wid  = t >> 5;
    const int lane = t & 31;
    const int lr   = lane & 15;
    const int lh   = lane >> 4;
    const int nBase = blockIdx.x * 128;
    const int mBase = blockIdx.y * 128;
    const int b     = blockIdx.z;
    const int wm = wid & 3;          // 0..3 -> 32-row slice
    const int wn = wid >> 2;         // 0..1 -> 64-col slice

    const size_t xBatch = ((size_t)b * xRows + xRowOff) * (size_t)Ntot;

    f32x8 acc[2][4];
    #pragma unroll
    for (int i = 0; i < 2; i++)
        #pragma unroll
        for (int j = 0; j < 4; j++) acc[i][j] = zero8();

    for (int k0 = 0; k0 < K; k0 += 32) {
        // --- stage W tile: 128 rows x 32 k, 16B per thread x2 ---
        #pragma unroll
        for (int it = 0; it < 2; it++) {
            int c   = t + it * 256;
            int row = c >> 2;
            int kc  = (c & 3) * 8;
            i32x4 d = *(const i32x4*)(Wm + (size_t)(mBase + row) * K + k0 + kc);
            *(i32x4*)&sW[row][kc] = d;
        }
        // --- stage X tile: 32 k x 128 n, coalesced read, transposed store ---
        #pragma unroll
        for (int it = 0; it < 2; it++) {
            int c  = t + it * 256;
            int kk = c >> 4;
            int nc = (c & 15) * 8;
            union { i32x4 q; bf16_t e[8]; } u;
            u.q = *(const i32x4*)(X + xBatch + (size_t)(k0 + kk) * Ntot + nBase + nc);
            #pragma unroll
            for (int j = 0; j < 8; j++) sX[nc + j][kk] = u.e[j];
        }
        __syncthreads();

        bf16x16 afr[2], bfr[4];
        #pragma unroll
        for (int tm = 0; tm < 2; tm++) {
            int row = wm * 32 + tm * 16 + lr;
            afr[tm] = load_frag(&sW[row][lh * 8], &sW[row][16 + lh * 8]);
        }
        #pragma unroll
        for (int tn = 0; tn < 4; tn++) {
            int col = wn * 64 + tn * 16 + lr;
            bfr[tn] = load_frag(&sX[col][lh * 8], &sX[col][16 + lh * 8]);
        }
        #pragma unroll
        for (int tm = 0; tm < 2; tm++)
            #pragma unroll
            for (int tn = 0; tn < 4; tn++)
                acc[tm][tn] = __builtin_amdgcn_wmma_f32_16x16x32_bf16(
                    false, afr[tm], false, bfr[tn], (short)0, acc[tm][tn], false, false);
        __syncthreads();
    }

    // --- epilogue: D tile lane map: M = j + lh*8, N = lr ---
    #pragma unroll
    for (int tm = 0; tm < 2; tm++) {
        #pragma unroll
        for (int j = 0; j < 8; j++) {
            int m = mBase + wm * 32 + tm * 16 + j + lh * 8;
            float bv = bias ? bias[m] : 0.0f;
            #pragma unroll
            for (int tn = 0; tn < 4; tn++) {
                int n = nBase + wn * 64 + tn * 16 + lr;
                float val = acc[tm][tn][j] + bv;
                size_t oi = ((size_t)b * oRows + oRowOff + m) * (size_t)Ntot + n;
                if (outF) outF[oi] = val;
                if (outB) outB[oi] = (bf16_t)val;
                if (outT) outT[((size_t)b * Ntot + n) * (size_t)M + m] = (bf16_t)val;
            }
        }
    }
}

// ---------------------------------------------------------------------------
// Flash attention: sim = softmax_m(q^T k); out[c][n] = gamma * sum_m sim * v[c][m]
// qT,kT: [b][n][P] bf16 (P=256).  vB: [b][C][n] bf16 (C=512).
// One workgroup = 16 queries; 8 waves; stream keys in blocks of 128.
// ---------------------------------------------------------------------------
__global__ __launch_bounds__(256) void attn_kernel(
    const bf16_t* __restrict__ qT, const bf16_t* __restrict__ kT,
    const bf16_t* __restrict__ vB, const float* __restrict__ gamma,
    float* __restrict__ osF, bf16_t* __restrict__ osB,
    int Ntot, int Cdim, int Pdim)
{
    __shared__ float  sS[16][128];
    __shared__ bf16_t sP[16][128];
    __shared__ float  sMax[16], sSum[16], sScale[16];

    const int t    = threadIdx.x;
    const int wid  = t >> 5;
    const int lane = t & 31;
    const int lr   = lane & 15;
    const int lh   = lane >> 4;
    const int qbase = blockIdx.x * 16;
    const int b     = blockIdx.y;

    if (t < 16) { sMax[t] = -1e30f; sSum[t] = 0.0f; }
    __syncthreads();

    const bf16_t* qTb = qT + (size_t)b * Ntot * Pdim;
    const bf16_t* kTb = kT + (size_t)b * Ntot * Pdim;
    const bf16_t* vBb = vB + (size_t)b * Cdim * Ntot;

    // resident Q fragments: 16 queries x full P=256 (8 K-chunks of 32)
    bf16x16 aq[8];
    #pragma unroll
    for (int kc = 0; kc < 8; kc++) {
        const bf16_t* rowp = qTb + (size_t)(qbase + lr) * Pdim + kc * 32;
        aq[kc] = load_frag(rowp + lh * 8, rowp + 16 + lh * 8);
    }

    f32x8 accO[4];  // this wave's 64-channel slice: 4 c-tiles of 16
    #pragma unroll
    for (int i = 0; i < 4; i++) accO[i] = zero8();

    for (int mb = 0; mb < Ntot; mb += 128) {
        // ---- phase 1: S tile; wave wid computes key columns [wid*16, wid*16+16) ----
        f32x8 accS = zero8();
        {
            const bf16_t* rowp = kTb + (size_t)(mb + wid * 16 + lr) * Pdim;
            #pragma unroll
            for (int kc = 0; kc < 8; kc++) {
                bf16x16 bk = load_frag(rowp + kc * 32 + lh * 8, rowp + kc * 32 + 16 + lh * 8);
                accS = __builtin_amdgcn_wmma_f32_16x16x32_bf16(
                    false, aq[kc], false, bk, (short)0, accS, false, false);
            }
        }
        #pragma unroll
        for (int j = 0; j < 8; j++) sS[j + lh * 8][wid * 16 + lr] = accS[j];
        __syncthreads();

        // ---- phase 2: online softmax; wave wid owns query rows 2*wid, 2*wid+1 ----
        #pragma unroll
        for (int rr = 0; rr < 2; rr++) {
            int r = wid * 2 + rr;
            float x0 = sS[r][lane], x1 = sS[r][lane + 32];
            float x2 = sS[r][lane + 64], x3 = sS[r][lane + 96];
            float ml = fmaxf(fmaxf(x0, x1), fmaxf(x2, x3));
            #pragma unroll
            for (int off = 16; off >= 1; off >>= 1) ml = fmaxf(ml, __shfl_xor(ml, off, 32));
            float mold = sMax[r];
            float mnew = fmaxf(mold, ml);
            float p0 = __expf(x0 - mnew), p1 = __expf(x1 - mnew);
            float p2 = __expf(x2 - mnew), p3 = __expf(x3 - mnew);
            sP[r][lane]      = (bf16_t)p0;
            sP[r][lane + 32] = (bf16_t)p1;
            sP[r][lane + 64] = (bf16_t)p2;
            sP[r][lane + 96] = (bf16_t)p3;
            float sl = p0 + p1 + p2 + p3;
            #pragma unroll
            for (int off = 16; off >= 1; off >>= 1) sl += __shfl_xor(sl, off, 32);
            if (lane == 0) {
                float sc  = __expf(mold - mnew);
                sMax[r]   = mnew;
                sScale[r] = sc;
                sSum[r]   = sSum[r] * sc + sl;
            }
        }
        __syncthreads();

        // ---- phase 3: rescale running output, then accO += P * V^T ----
        float fs[8];
        #pragma unroll
        for (int j = 0; j < 8; j++) fs[j] = sScale[j + lh * 8];
        #pragma unroll
        for (int tc = 0; tc < 4; tc++)
            #pragma unroll
            for (int j = 0; j < 8; j++) accO[tc][j] *= fs[j];

        bf16x16 ap[4];
        #pragma unroll
        for (int mc = 0; mc < 4; mc++)
            ap[mc] = load_frag(&sP[lr][mc * 32 + lh * 8], &sP[lr][mc * 32 + 16 + lh * 8]);

        #pragma unroll
        for (int tc = 0; tc < 4; tc++) {
            int ccol = wid * 64 + tc * 16 + lr;
            const bf16_t* vrow = vBb + (size_t)ccol * Ntot + mb;
            #pragma unroll
            for (int mc = 0; mc < 4; mc++) {
                bf16x16 bv = load_frag(vrow + mc * 32 + lh * 8, vrow + mc * 32 + 16 + lh * 8);
                accO[tc] = __builtin_amdgcn_wmma_f32_16x16x32_bf16(
                    false, ap[mc], false, bv, (short)0, accO[tc], false, false);
            }
        }
        __syncthreads();
    }

    // ---- epilogue: divide by rowsum, scale by gamma, store f32 + bf16 ----
    float g = *gamma;
    float fi[8];
    #pragma unroll
    for (int j = 0; j < 8; j++) fi[j] = g / sSum[j + lh * 8];
    #pragma unroll
    for (int tc = 0; tc < 4; tc++) {
        int ccol = wid * 64 + tc * 16 + lr;
        #pragma unroll
        for (int j = 0; j < 8; j++) {
            int n = qbase + j + lh * 8;
            float val = accO[tc][j] * fi[j];
            size_t oi = ((size_t)b * Cdim + ccol) * (size_t)Ntot + n;
            osF[oi] = val;
            osB[oi] = (bf16_t)val;
        }
    }
}

// ---------------------------------------------------------------------------
// Final: sigmoid of H4..H6, softmax across the 3, residual combine.
// ---------------------------------------------------------------------------
__global__ void combine_kernel(
    const float* __restrict__ x,
    const float* __restrict__ h0, const float* __restrict__ h1, const float* __restrict__ h2,
    const float* __restrict__ o0, const float* __restrict__ o1, const float* __restrict__ o2,
    float* __restrict__ out, int total)
{
    int i = blockIdx.x * blockDim.x + threadIdx.x;
    if (i >= total) return;
    float s0 = 1.0f / (1.0f + __expf(-h0[i]));
    float s1 = 1.0f / (1.0f + __expf(-h1[i]));
    float s2 = 1.0f / (1.0f + __expf(-h2[i]));
    float mx = fmaxf(s0, fmaxf(s1, s2));
    float e0 = __expf(s0 - mx), e1 = __expf(s1 - mx), e2 = __expf(s2 - mx);
    float inv = 1.0f / (e0 + e1 + e2);
    out[i] = x[i] + (e0 * o0[i] + e1 * o1[i] + e2 * o2[i]) * inv;
}

// ---------------------------------------------------------------------------
extern "C" void kernel_launch(void* const* d_in, const int* in_sizes, int n_in,
                              void* d_out, int out_size, void* d_ws, size_t ws_size,
                              hipStream_t stream)
{
    (void)in_sizes; (void)n_in; (void)out_size; (void)ws_size;
    constexpr int Bn = 2, Cc = 512, Nn = 4096, Pp = 256, C3 = 1536;

    const float* X    = (const float*)d_in[0];
    const float* Y    = (const float*)d_in[1];
    const float* Z    = (const float*)d_in[2];
    const float* Wv   = (const float*)d_in[3];
    const float* Wv1  = (const float*)d_in[4];
    const float* Wv2  = (const float*)d_in[5];
    const float* Wq   = (const float*)d_in[6];
    const float* bq   = (const float*)d_in[7];
    const float* Wk   = (const float*)d_in[8];
    const float* bk   = (const float*)d_in[9];
    const float* Wq1  = (const float*)d_in[10];
    const float* bq1  = (const float*)d_in[11];
    const float* Wk1  = (const float*)d_in[12];
    const float* bk1  = (const float*)d_in[13];
    const float* Wq2  = (const float*)d_in[14];
    const float* bq2  = (const float*)d_in[15];
    const float* Wk2  = (const float*)d_in[16];
    const float* bk2  = (const float*)d_in[17];
    const float* Win1 = (const float*)d_in[18];
    const float* bin1 = (const float*)d_in[19];
    const float* Win2 = (const float*)d_in[20];
    const float* bin2 = (const float*)d_in[21];
    const float* Win3 = (const float*)d_in[22];
    const float* bin3 = (const float*)d_in[23];
    const float* Wtr  = (const float*)d_in[24];
    const float* btr  = (const float*)d_in[25];
    const float* Wout1= (const float*)d_in[26];
    const float* bout1= (const float*)d_in[27];
    const float* Wout2= (const float*)d_in[28];
    const float* bout2= (const float*)d_in[29];
    const float* Wout3= (const float*)d_in[30];
    const float* bout3= (const float*)d_in[31];
    const float* gam0 = (const float*)d_in[32];
    const float* gam1 = (const float*)d_in[33];
    const float* gam2 = (const float*)d_in[34];

    char* wp = (char*)d_ws;
    size_t off = 0;
    auto alloc = [&](size_t bytes) -> void* {
        void* p = wp + off;
        off += (bytes + 255) & ~(size_t)255;
        return p;
    };

    const size_t actE = (size_t)Bn * Cc * Nn;       // per-activation elements
    bf16_t* xb = (bf16_t*)alloc(actE * 2);
    bf16_t* yb = (bf16_t*)alloc(actE * 2);
    bf16_t* zb = (bf16_t*)alloc(actE * 2);
    bf16_t* wvb[3], *wqb[3], *wkb[3], *winb[3], *woutb[3];
    for (int i = 0; i < 3; i++) wvb[i]   = (bf16_t*)alloc((size_t)Cc * Cc * 2);
    for (int i = 0; i < 3; i++) { wqb[i] = (bf16_t*)alloc((size_t)Pp * Cc * 2);
                                  wkb[i] = (bf16_t*)alloc((size_t)Pp * Cc * 2); }
    for (int i = 0; i < 3; i++) winb[i]  = (bf16_t*)alloc((size_t)Cc * Cc * 2);
    bf16_t* wtrb = (bf16_t*)alloc((size_t)C3 * C3 * 2);
    for (int i = 0; i < 3; i++) woutb[i] = (bf16_t*)alloc((size_t)Cc * Cc * 2);

    bf16_t *val[3], *qTb[3], *kTb[3], *osB[3];
    float  *osF[3], *hpre[3];
    for (int i = 0; i < 3; i++) val[i] = (bf16_t*)alloc(actE * 2);
    for (int i = 0; i < 3; i++) { qTb[i] = (bf16_t*)alloc((size_t)Bn * Nn * Pp * 2);
                                  kTb[i] = (bf16_t*)alloc((size_t)Bn * Nn * Pp * 2); }
    for (int i = 0; i < 3; i++) osF[i] = (float*)alloc(actE * 4);
    for (int i = 0; i < 3; i++) osB[i] = (bf16_t*)alloc(actE * 2);
    bf16_t* hcat = (bf16_t*)alloc((size_t)Bn * C3 * Nn * 2);
    bf16_t* strB = (bf16_t*)alloc((size_t)Bn * C3 * Nn * 2);
    for (int i = 0; i < 3; i++) hpre[i] = (float*)alloc(actE * 4);

    auto cvt = [&](const float* s, bf16_t* d, int n) {
        cvt_kernel<<<(n + 255) / 256, 256, 0, stream>>>(s, d, n);
    };
    cvt(X, xb, (int)actE); cvt(Y, yb, (int)actE); cvt(Z, zb, (int)actE);
    cvt(Wv, wvb[0], Cc * Cc); cvt(Wv1, wvb[1], Cc * Cc); cvt(Wv2, wvb[2], Cc * Cc);
    cvt(Wq, wqb[0], Pp * Cc);  cvt(Wk, wkb[0], Pp * Cc);
    cvt(Wq1, wqb[1], Pp * Cc); cvt(Wk1, wkb[1], Pp * Cc);
    cvt(Wq2, wqb[2], Pp * Cc); cvt(Wk2, wkb[2], Pp * Cc);
    cvt(Win1, winb[0], Cc * Cc); cvt(Win2, winb[1], Cc * Cc); cvt(Win3, winb[2], Cc * Cc);
    cvt(Wtr, wtrb, C3 * C3);
    cvt(Wout1, woutb[0], Cc * Cc); cvt(Wout2, woutb[1], Cc * Cc); cvt(Wout3, woutb[2], Cc * Cc);

    auto gemm = [&](const bf16_t* Wm, const bf16_t* Xp, const float* bias,
                    float* oF, bf16_t* oB, bf16_t* oT,
                    int M, int K, int xRows, int xRowOff, int oRows, int oRowOff) {
        dim3 grid(Nn / 128, M / 128, Bn);
        wmma_gemm<<<grid, 256, 0, stream>>>(Wm, Xp, bias, oF, oB, oT,
                                            M, Nn, K, xRows, xRowOff, oRows, oRowOff);
    };

    // projections: values (natural bf16) and q/k (transposed [n][p] bf16, with bias)
    gemm(wvb[0], yb, nullptr, nullptr, val[0], nullptr, Cc, Cc, Cc, 0, Cc, 0);
    gemm(wvb[1], zb, nullptr, nullptr, val[1], nullptr, Cc, Cc, Cc, 0, Cc, 0);
    gemm(wvb[2], xb, nullptr, nullptr, val[2], nullptr, Cc, Cc, Cc, 0, Cc, 0);
    gemm(wqb[0], xb, bq,  nullptr, nullptr, qTb[0], Pp, Cc, Cc, 0, Pp, 0);
    gemm(wkb[0], yb, bk,  nullptr, nullptr, kTb[0], Pp, Cc, Cc, 0, Pp, 0);
    gemm(wqb[1], xb, bq1, nullptr, nullptr, qTb[1], Pp, Cc, Cc, 0, Pp, 0);
    gemm(wkb[1], zb, bk1, nullptr, nullptr, kTb[1], Pp, Cc, Cc, 0, Pp, 0);
    gemm(wqb[2], xb, bq2, nullptr, nullptr, qTb[2], Pp, Cc, Cc, 0, Pp, 0);
    gemm(wkb[2], xb, bk2, nullptr, nullptr, kTb[2], Pp, Cc, Cc, 0, Pp, 0);

    // attention blocks (flash-style)
    dim3 ag(Nn / 16, Bn);
    const float* gs[3] = {gam0, gam1, gam2};
    for (int i = 0; i < 3; i++)
        attn_kernel<<<ag, 256, 0, stream>>>(qTb[i], kTb[i], val[i], gs[i],
                                            osF[i], osB[i], Nn, Cc, Pp);

    // H1..H3 -> concatenated [b][1536][n] bf16
    const float* bins[3] = {bin1, bin2, bin3};
    for (int i = 0; i < 3; i++)
        gemm(winb[i], osB[i], bins[i], nullptr, hcat, nullptr, Cc, Cc, Cc, 0, C3, i * Cc);

    // H_tra = Wtr (1536x1536) @ hcat + btr
    gemm(wtrb, hcat, btr, nullptr, strB, nullptr, C3, C3, C3, 0, C3, 0);

    // H4..H6 pre-sigmoid (f32) from s1..s3 slices of H_tra
    const float* bouts[3] = {bout1, bout2, bout3};
    for (int i = 0; i < 3; i++)
        gemm(woutb[i], strB, bouts[i], hpre[i], nullptr, nullptr, Cc, Cc, C3, i * Cc, Cc, 0);

    // final combine
    int total = (int)actE;
    combine_kernel<<<(total + 255) / 256, 256, 0, stream>>>(
        X, hpre[0], hpre[1], hpre[2], osF[0], osF[1], osF[2], (float*)d_out, total);
}
